// ScenarioExtractionLayer_86955907875168
// MI455X (gfx1250) — matrix operand order repaired and checked by reading, not compiled, not run on hardware
//
#include <hip/hip_runtime.h>
#include <hip/hip_bf16.h>
#include <stdint.h>

typedef __bf16 bf16;
typedef __attribute__((ext_vector_type(16))) __bf16 v16bf;
typedef __attribute__((ext_vector_type(8)))  float  v8f;
typedef __attribute__((ext_vector_type(4)))  unsigned int u32x4;

#define B_  16384
#define D_  512
#define H_  1024
#define S_  8
#define E_  4
#define DH_ 256

// ---------------- fp32 -> bf16 convert ----------------
__global__ void k_f32_to_bf16(const float* __restrict__ src, bf16* __restrict__ dst, int n) {
  int i = blockIdx.x * blockDim.x + threadIdx.x;
  int stride = gridDim.x * blockDim.x;
  for (; i < n; i += stride) dst[i] = (bf16)src[i];
}

// ---------------- gate softmax: gates[b,e] = softmax(x@gate_w + gate_b) ----------------
__global__ __launch_bounds__(256) void k_gate(const float* __restrict__ x,
                                              const float* __restrict__ gw,
                                              const float* __restrict__ gb,
                                              float* __restrict__ gates) {
  int wave = threadIdx.x >> 5, lane = threadIdx.x & 31;
  int b = blockIdx.x * 8 + wave;
  const float* xr = x + (long)b * D_;
  float acc[E_] = {0.f, 0.f, 0.f, 0.f};
  for (int d = lane; d < D_; d += 32) {
    float xv = xr[d];
    const float* g = gw + d * E_;
#pragma unroll
    for (int e = 0; e < E_; e++) acc[e] += xv * g[e];
  }
#pragma unroll
  for (int e = 0; e < E_; e++) {
    float a = acc[e];
#pragma unroll
    for (int off = 16; off >= 1; off >>= 1) a += __shfl_xor(a, off);
    acc[e] = a + gb[e];
  }
  float m = fmaxf(fmaxf(acc[0], acc[1]), fmaxf(acc[2], acc[3]));
  float p[E_]; float s = 0.f;
#pragma unroll
  for (int e = 0; e < E_; e++) { p[e] = __expf(acc[e] - m); s += p[e]; }
  float inv = 1.f / s;
  if (lane == 0) {
#pragma unroll
    for (int e = 0; e < E_; e++) gates[b * E_ + e] = p[e] * inv;
  }
}

// ---------------- counting sort by scenario (single block) ----------------
__global__ void k_sort(const int* __restrict__ sid, int* __restrict__ order,
                       int* __restrict__ offsets) {
  __shared__ int cnt[S_], cur[S_];
  int t = threadIdx.x;
  if (t < S_) cnt[t] = 0;
  __syncthreads();
  for (int i = t; i < B_; i += blockDim.x) atomicAdd(&cnt[sid[i]], 1);
  __syncthreads();
  if (t == 0) {
    int o = 0;
    for (int s = 0; s < S_; s++) { offsets[s] = o; cur[s] = o; o += cnt[s]; }
    offsets[S_] = o;
  }
  __syncthreads();
  for (int i = t; i < B_; i += blockDim.x) {
    int pos = atomicAdd(&cur[sid[i]], 1);
    order[pos] = i;
  }
}

// ---------------- scen_attn[s,j] = scen_embed[s,:] @ attn_w1[D:,j] + attn_b1[j] ----------------
__global__ void k_scen_attn(const float* __restrict__ scen_embed,
                            const float* __restrict__ attn_w1,
                            const float* __restrict__ attn_b1,
                            float* __restrict__ scen_attn) {
  int s = blockIdx.x, j = threadIdx.x;
  float acc = attn_b1[j];
  const float* se = scen_embed + s * D_;
  for (int d = 0; d < D_; d++) acc += se[d] * attn_w1[(long)(D_ + d) * DH_ + j];
  scen_attn[s * DH_ + j] = acc;
}

// ---------------- per-row: relu + w2 dot + softmax over S, select wsel ----------------
__global__ __launch_bounds__(256) void k_attn_finish(const float* __restrict__ sh_attn,
                                                     const float* __restrict__ scen_attn,
                                                     const float* __restrict__ attn_w2,
                                                     const float* __restrict__ attn_b2,
                                                     const int* __restrict__ sid,
                                                     float* __restrict__ wsel) {
  int wave = threadIdx.x >> 5, lane = threadIdx.x & 31;
  int b = blockIdx.x * 8 + wave;
  float sh[8], w2[8];
#pragma unroll
  for (int i = 0; i < 8; i++) {
    sh[i] = sh_attn[(long)b * DH_ + lane + 32 * i];
    w2[i] = attn_w2[lane + 32 * i];
  }
  float logit[S_];
#pragma unroll
  for (int s = 0; s < S_; s++) {
    float acc = 0.f;
#pragma unroll
    for (int i = 0; i < 8; i++)
      acc += fmaxf(sh[i] + scen_attn[s * DH_ + lane + 32 * i], 0.f) * w2[i];
#pragma unroll
    for (int off = 16; off >= 1; off >>= 1) acc += __shfl_xor(acc, off);
    logit[s] = acc + attn_b2[0];
  }
  float m = logit[0];
#pragma unroll
  for (int s = 1; s < S_; s++) m = fmaxf(m, logit[s]);
  float sum = 0.f;
#pragma unroll
  for (int s = 0; s < S_; s++) sum += __expf(logit[s] - m);
  int sb = sid[b];
  float p = __expf(logit[sb] - m) / sum;
  if (lane == 0) wsel[b] = p;
}

// ---------------- generic bf16 WMMA GEMM ----------------
// Block tile 128(M) x 64(N), BK=32. 8 waves as 4(M) x 2(N); each wave computes a
// 32x32 output via 4x v_wmma_f32_16x16x32_bf16 per K-step (2 A-frags x 2 B-frags).
// A tile staged with async global->LDS (CDNA5 GLOBAL_LOAD_ASYNC_TO_LDS_B128).
__global__ __launch_bounds__(256) void k_gemm(
    const bf16* __restrict__ A, int lda,
    const bf16* __restrict__ Bw, int ldb, long b_stride,
    const float* __restrict__ bias, int bias_stride,
    float* __restrict__ outF, bf16* __restrict__ outB, int ldc, int accum, int relu,
    int M, int N, int K,
    const int* __restrict__ in_map, const int* __restrict__ out_map,
    const int* __restrict__ group_off,
    const float* __restrict__ post_scale, int ps_stride,
    const float* __restrict__ bias_scale, int bs_stride,
    const bf16* __restrict__ enh_sh, const bf16* __restrict__ enh_sp,
    const float* __restrict__ enh_w, int enh_mode)
{
  __shared__ bf16 As[128 * 40];  // [row][k], row stride 40 elems (80B, 16B aligned)
  __shared__ bf16 BsT[64 * 40];  // [col][k], col stride 40 elems

  int z = blockIdx.z;
  int base = 0, count = M;
  if (group_off) { base = group_off[z]; count = group_off[z + 1] - base; }
  int m0 = blockIdx.x * 128;
  if (m0 >= count) return;
  int n0 = blockIdx.y * 64;

  const bf16* Bp = Bw + (long)z * b_stride;
  const float* biasp = bias ? bias + (long)z * bias_stride : nullptr;

  int tid = threadIdx.x;
  // A loader: 128 rows x 2 groups of 16 cols. Beyond-count rows clamp to the
  // group's first row (their C rows are masked at store; rows are independent).
  int arow_l = tid >> 1;
  int acg = (tid & 1) * 16;
  int aig = m0 + arow_l;
  bool avalid = aig < count;
  int asrc = base + (avalid ? aig : 0);
  int a_row = in_map ? in_map[asrc] : asrc;
  float wv = enh_mode ? enh_w[a_row] : 0.f;
  unsigned a_lds = (unsigned)(uintptr_t)(&As[arow_l * 40 + acg]);  // LDS byte offset
  // B loader: 32 K-rows x 8 groups of 8 cols
  int brow = tid >> 3;
  int bcg = (tid & 7) * 8;

  int wave = tid >> 5, lane = tid & 31;
  int wm = wave & 3, wn = wave >> 2;
  int lrow = lane & 15, khalf = lane >> 4;

  v8f c00 = {0.f, 0.f, 0.f, 0.f, 0.f, 0.f, 0.f, 0.f};
  v8f c01 = c00, c10 = c00, c11 = c00;

  for (int k0 = 0; k0 < K; k0 += 32) {
    // ---- stage A tile (128x32) ----
    if (!enh_mode) {
      const bf16* src = A + (long)a_row * lda + k0 + acg;
      // async copy 32B: global -> LDS, offset applies to both addresses
      asm volatile("global_load_async_to_lds_b128 %0, %1, off"
                   :: "v"(a_lds), "v"(src) : "memory");
      asm volatile("global_load_async_to_lds_b128 %0, %1, off offset:16"
                   :: "v"(a_lds), "v"(src) : "memory");
    } else {
      // enhanced = [shared, w*shared, specific, w*specific]; 16-elem chunk stays
      // inside one 512-wide segment (k0%32==0, acg%16==0).
      int kg = k0 + acg;
      int seg = kg >> 9;
      int kk = kg & (D_ - 1);
      const bf16* src = ((seg < 2) ? enh_sh : enh_sp) + (long)a_row * D_ + kk;
      float mul = (seg & 1) ? wv : 1.f;
#pragma unroll
      for (int j = 0; j < 16; j++)
        As[arow_l * 40 + acg + j] = (bf16)((float)src[j] * mul);
    }
    // ---- stage B tile transposed (BsT[col][k], 64 cols x 32 k) ----
    {
      u32x4 v = *(const u32x4*)(Bp + (long)(k0 + brow) * ldb + n0 + bcg);
      const bf16* pv = (const bf16*)&v;
#pragma unroll
      for (int j = 0; j < 8; j++) BsT[(bcg + j) * 40 + brow] = pv[j];
    }
    if (!enh_mode) {
      asm volatile("s_wait_asynccnt 0x0" ::: "memory");
    }
    __syncthreads();

    // ---- build fragments per ISA VGPR layouts ----
    union { u32x4 q[2]; v16bf v; } ua0, ua1, ub0, ub1;
    // A 16x32: lanes 0-15 / 16-31: j<8 -> K=8*khalf+j ; j>=8 -> K=16+8*khalf+(j-8)
    int r0 = (wm * 32 + lrow) * 40;
    int r1 = (wm * 32 + 16 + lrow) * 40;
    ua0.q[0] = *(const u32x4*)&As[r0 + khalf * 8];
    ua0.q[1] = *(const u32x4*)&As[r0 + 16 + khalf * 8];
    ua1.q[0] = *(const u32x4*)&As[r1 + khalf * 8];
    ua1.q[1] = *(const u32x4*)&As[r1 + 16 + khalf * 8];
    // B 32x16: lane holds col n=lane%16; j<8 -> K=8*khalf+j ; j>=8 -> K=16+8*khalf+(j-8)
    int nc0 = wn * 32 + lrow;
    ub0.q[0] = *(const u32x4*)&BsT[nc0 * 40 + khalf * 8];
    ub0.q[1] = *(const u32x4*)&BsT[nc0 * 40 + 16 + khalf * 8];
    ub1.q[0] = *(const u32x4*)&BsT[(nc0 + 16) * 40 + khalf * 8];
    ub1.q[1] = *(const u32x4*)&BsT[(nc0 + 16) * 40 + 16 + khalf * 8];

    c00 = __builtin_amdgcn_wmma_f32_16x16x32_bf16(false, ua0.v, false, ub0.v,
                                                  (short)0, c00, false, false);
    c01 = __builtin_amdgcn_wmma_f32_16x16x32_bf16(false, ua0.v, false, ub1.v,
                                                  (short)0, c01, false, false);
    c10 = __builtin_amdgcn_wmma_f32_16x16x32_bf16(false, ua1.v, false, ub0.v,
                                                  (short)0, c10, false, false);
    c11 = __builtin_amdgcn_wmma_f32_16x16x32_bf16(false, ua1.v, false, ub1.v,
                                                  (short)0, c11, false, false);
    __syncthreads();
  }

  // ---- epilogue: C 16x16 f32 layout: VGPR v -> M = v + 8*khalf, N = lane%16 ----
#pragma unroll
  for (int sub = 0; sub < 2; sub++) {
#pragma unroll
    for (int t = 0; t < 2; t++) {
      int n = n0 + wn * 32 + lrow + t * 16;
      float bv = biasp ? biasp[n] : 0.f;
      v8f cc = sub ? (t ? c11 : c10) : (t ? c01 : c00);
#pragma unroll
      for (int v = 0; v < 8; v++) {
        int ml = wm * 32 + sub * 16 + v + (khalf ? 8 : 0);
        int ig = m0 + ml;
        if (ig < count) {
          int crow = out_map ? out_map[base + ig] : (base + ig);
          float bscale = bias_scale ? bias_scale[(long)crow * bs_stride] : 1.f;
          float val = cc[v] + bv * bscale;
          if (relu) val = fmaxf(val, 0.f);
          if (post_scale) val *= post_scale[(long)crow * ps_stride];
          long idx = (long)crow * ldc + n;
          if (outF) outF[idx] = accum ? (outF[idx] + val) : val;
          else      outB[idx] = (bf16)val;
        }
      }
    }
  }
}

extern "C" void kernel_launch(void* const* d_in, const int* in_sizes, int n_in,
                              void* d_out, int out_size, void* d_ws, size_t ws_size,
                              hipStream_t stream) {
  (void)in_sizes; (void)n_in; (void)out_size; (void)ws_size;
  const float* x          = (const float*)d_in[0];
  const int*   sid        = (const int*)  d_in[1];
  const float* gate_w     = (const float*)d_in[2];
  const float* gate_b     = (const float*)d_in[3];
  const float* sub_w1     = (const float*)d_in[4];
  const float* sub_b1     = (const float*)d_in[5];
  const float* sub_w2     = (const float*)d_in[6];
  const float* sub_b2     = (const float*)d_in[7];
  const float* spec_w1    = (const float*)d_in[8];
  const float* spec_b1    = (const float*)d_in[9];
  const float* spec_w2    = (const float*)d_in[10];
  const float* spec_b2    = (const float*)d_in[11];
  const float* scen_embed = (const float*)d_in[12];
  const float* attn_w1    = (const float*)d_in[13];
  const float* attn_b1    = (const float*)d_in[14];
  const float* attn_w2    = (const float*)d_in[15];
  const float* attn_b2    = (const float*)d_in[16];
  const float* out_w      = (const float*)d_in[17];
  const float* out_b      = (const float*)d_in[18];
  float* out = (float*)d_out;

  char* ws = (char*)d_ws;
  size_t off = 0;
  auto alloc = [&](size_t bytes) -> char* {
    char* p = ws + off;
    off = (off + bytes + 255) & ~(size_t)255;
    return p;
  };
  bf16*  x_bf        = (bf16*) alloc((size_t)B_ * D_ * 2);
  bf16*  sub_w1_bf   = (bf16*) alloc((size_t)E_ * D_ * H_ * 2);
  bf16*  sub_w2_bf   = (bf16*) alloc((size_t)E_ * H_ * D_ * 2);
  bf16*  spec_w1_bf  = (bf16*) alloc((size_t)S_ * D_ * H_ * 2);
  bf16*  spec_w2_bf  = (bf16*) alloc((size_t)S_ * H_ * D_ * 2);
  bf16*  attn_w1a_bf = (bf16*) alloc((size_t)D_ * DH_ * 2);
  bf16*  out_w_bf    = (bf16*) alloc((size_t)4 * D_ * D_ * 2);
  float* gates       = (float*)alloc((size_t)B_ * E_ * 4);
  bf16*  h_bf        = (bf16*) alloc((size_t)B_ * H_ * 2);
  float* shared_f    = (float*)alloc((size_t)B_ * D_ * 4);
  bf16*  shared_bf   = (bf16*) alloc((size_t)B_ * D_ * 2);
  bf16*  spec_bf     = (bf16*) alloc((size_t)B_ * D_ * 2);
  float* sh_attn     = (float*)alloc((size_t)B_ * DH_ * 4);
  float* scen_attn   = (float*)alloc((size_t)S_ * DH_ * 4);
  float* wsel        = (float*)alloc((size_t)B_ * 4);
  int*   order       = (int*)  alloc((size_t)B_ * 4);
  int*   offsets     = (int*)  alloc((size_t)(S_ + 1) * 4);

  auto cv = [&](const float* s, bf16* d, long n) {
    int blocks = (int)((n + 255) / 256);
    if (blocks > 2048) blocks = 2048;
    k_f32_to_bf16<<<blocks, 256, 0, stream>>>(s, d, (int)n);
  };
  cv(x,       x_bf,        (long)B_ * D_);
  cv(sub_w1,  sub_w1_bf,   (long)E_ * D_ * H_);
  cv(sub_w2,  sub_w2_bf,   (long)E_ * H_ * D_);
  cv(spec_w1, spec_w1_bf,  (long)S_ * D_ * H_);
  cv(spec_w2, spec_w2_bf,  (long)S_ * H_ * D_);
  cv(attn_w1, attn_w1a_bf, (long)D_ * DH_);      // first D rows only
  cv(out_w,   out_w_bf,    (long)4 * D_ * D_);

  k_gate<<<B_ / 8, 256, 0, stream>>>(x, gate_w, gate_b, gates);
  k_sort<<<1, 256, 0, stream>>>(sid, order, offsets);

  dim3 blk(256);
  const int MT = B_ / 128;   // M tiles
  // ---- shared experts: h = g[b,e]*relu(x@w1+b1); shared += h@w2 + g[b,e]*b2 ----
  for (int e = 0; e < E_; e++) {
    k_gemm<<<dim3(MT, H_ / 64, 1), blk, 0, stream>>>(
        x_bf, D_, sub_w1_bf + (long)e * D_ * H_, H_, 0,
        sub_b1 + e * H_, 0,
        nullptr, h_bf, H_, 0, 1,
        B_, H_, D_,
        nullptr, nullptr, nullptr,
        gates + e, E_,            // post-scale by gate
        nullptr, 0,
        nullptr, nullptr, nullptr, 0);
    k_gemm<<<dim3(MT, D_ / 64, 1), blk, 0, stream>>>(
        h_bf, H_, sub_w2_bf + (long)e * H_ * D_, D_, 0,
        sub_b2 + e * D_, 0,
        shared_f, nullptr, D_, (e > 0) ? 1 : 0, 0,
        B_, D_, H_,
        nullptr, nullptr, nullptr,
        nullptr, 0,
        gates + e, E_,            // bias scaled by gate
        nullptr, nullptr, nullptr, 0);
  }
  // ---- scenario experts on grouped rows only ----
  k_gemm<<<dim3(MT, H_ / 64, S_), blk, 0, stream>>>(
      x_bf, D_, spec_w1_bf, H_, (long)D_ * H_,
      spec_b1, H_,
      nullptr, h_bf, H_, 0, 1,
      B_, H_, D_,
      order, nullptr, offsets,    // gather rows, store compact
      nullptr, 0, nullptr, 0,
      nullptr, nullptr, nullptr, 0);
  k_gemm<<<dim3(MT, D_ / 64, S_), blk, 0, stream>>>(
      h_bf, H_, spec_w2_bf, D_, (long)H_ * D_,
      spec_b2, D_,
      nullptr, spec_bf, D_, 0, 0,
      B_, D_, H_,
      nullptr, order, offsets,    // compact rows, scatter to original b
      nullptr, 0, nullptr, 0,
      nullptr, nullptr, nullptr, 0);

  cv(shared_f, shared_bf, (long)B_ * D_);

  // ---- attention: shared part via WMMA, scenario part + finish scalar ----
  k_gemm<<<dim3(MT, DH_ / 64, 1), blk, 0, stream>>>(
      shared_bf, D_, attn_w1a_bf, DH_, 0,
      nullptr, 0,
      sh_attn, nullptr, DH_, 0, 0,
      B_, DH_, D_,
      nullptr, nullptr, nullptr,
      nullptr, 0, nullptr, 0,
      nullptr, nullptr, nullptr, 0);
  k_scen_attn<<<S_, DH_, 0, stream>>>(scen_embed, attn_w1, attn_b1, scen_attn);
  k_attn_finish<<<B_ / 8, 256, 0, stream>>>(sh_attn, scen_attn, attn_w2, attn_b2, sid, wsel);

  // ---- out = [shared, w*shared, specific, w*specific] @ out_w + out_b ----
  k_gemm<<<dim3(MT, D_ / 64, 1), blk, 0, stream>>>(
      nullptr, 0, out_w_bf, D_, 0,
      out_b, 0,
      out, nullptr, D_, 0, 0,
      B_, D_, 4 * D_,
      nullptr, nullptr, nullptr,
      nullptr, 0, nullptr, 0,
      shared_bf, spec_bf, wsel, 1);
}
